// CPCLossV2_81286551044366
// MI455X (gfx1250) — matrix operand order repaired and checked by reading, not compiled, not run on hardware
//
#include <hip/hip_runtime.h>
#include <hip/hip_bf16.h>

typedef __attribute__((ext_vector_type(16))) __bf16 v16bf;
typedef __attribute__((ext_vector_type(8)))  float  v8f;

#define NGRP 4096
#define HID  256
#define KDIM 768          // (k-1)*h
#define MNEG 256

union V16 { v16bf v; uint4 q[2]; };

__device__ __forceinline__ void split2(float x, __bf16& hi, __bf16& lo) {
    __bf16 h = (__bf16)x;
    hi = h;
    lo = (__bf16)(x - (float)h);
}

// ---------------- fp32 -> (bf16 hi, bf16 lo) precompute ----------------
__global__ void __launch_bounds__(256)
cpc_split_kernel(const float* __restrict__ src,
                 __bf16* __restrict__ hi, __bf16* __restrict__ lo, int n)
{
    int i = blockIdx.x * 256 + threadIdx.x;
    if (i < n) {
        __bf16 h, l;
        split2(src[i], h, l);
        hi[i] = h;
        lo[i] = l;
    }
}

// ---------------- GEMM v2: bf16 fragments loaded directly ----------------
// predicts[4096][256] = hist_x[4096][768] * W^T + b, bf16x3 emulated fp32.
__global__ void __launch_bounds__(256)
cpc_gemm_bf16_kernel(const __bf16* __restrict__ ehi, const __bf16* __restrict__ elo,
                     const __bf16* __restrict__ whi, const __bf16* __restrict__ wlo,
                     const float* __restrict__ bias, float* __restrict__ predicts)
{
    const int lane      = threadIdx.x & 31;
    const int wave      = blockIdx.x * 8 + (threadIdx.x >> 5); // 1024 waves
    const int row_tile  = wave >> 2;      // 0..255
    const int col_block = wave & 3;       // 0..3 (64 cols each)
    const int mrow  = lane & 15;
    const int khalf = lane >> 4;

    const size_t arow = (size_t)(row_tile * 16 + mrow) * 1024; // hist_x row base

    v8f acc[4] = {};

    for (int kb = 0; kb < KDIM; kb += 32) {
        // A fragment: elems 0..7 = K kb+8*khalf.., elems 8..15 = K kb+16+8*khalf..
        V16 ah, al;
        const __bf16* a0 = ehi + arow + kb + khalf * 8;
        const __bf16* a1 = elo + arow + kb + khalf * 8;
        ah.q[0] = *(const uint4*)(a0);
        ah.q[1] = *(const uint4*)(a0 + 16);
        al.q[0] = *(const uint4*)(a1);
        al.q[1] = *(const uint4*)(a1 + 16);

        #pragma unroll
        for (int t = 0; t < 4; ++t) {
            const int j = col_block * 64 + t * 16 + mrow;       // output col = W row
            const size_t boff = (size_t)j * KDIM + kb + khalf * 8;
            V16 bh, bl;
            const __bf16* b0 = whi + boff;
            const __bf16* b1 = wlo + boff;
            bh.q[0] = *(const uint4*)(b0);
            bh.q[1] = *(const uint4*)(b0 + 16);
            bl.q[0] = *(const uint4*)(b1);
            bl.q[1] = *(const uint4*)(b1 + 16);

            acc[t] = __builtin_amdgcn_wmma_f32_16x16x32_bf16(false, ah.v, false, bh.v, (short)0, acc[t], false, false);
            acc[t] = __builtin_amdgcn_wmma_f32_16x16x32_bf16(false, ah.v, false, bl.v, (short)0, acc[t], false, false);
            acc[t] = __builtin_amdgcn_wmma_f32_16x16x32_bf16(false, al.v, false, bh.v, (short)0, acc[t], false, false);
        }
    }

    const int rowbase = row_tile * 16 + khalf * 8;
    #pragma unroll
    for (int t = 0; t < 4; ++t) {
        const int col = col_block * 64 + t * 16 + mrow;
        const float bv = bias[col];
        #pragma unroll
        for (int r = 0; r < 8; ++r)
            predicts[(size_t)(rowbase + r) * HID + col] = acc[t][r] + bv;
    }
}

// ---------------- GEMM fallback: convert fp32 -> bf16 hi/lo in-loop ----------------
__global__ void __launch_bounds__(256)
cpc_gemm_fallback_kernel(const float* __restrict__ emb,
                         const float* __restrict__ Wm,
                         const float* __restrict__ bias,
                         float* __restrict__ predicts)
{
    const int lane      = threadIdx.x & 31;
    const int wave      = blockIdx.x * 8 + (threadIdx.x >> 5);
    const int row_tile  = wave >> 2;
    const int col_block = wave & 3;
    const int mrow  = lane & 15;
    const int khalf = lane >> 4;

    const float* __restrict__ Arow = emb + (size_t)(row_tile * 16 + mrow) * 1024;

    v8f acc[4] = {};

    for (int kb = 0; kb < KDIM; kb += 32) {
        float af[16];
        const float* ap  = Arow + kb + khalf * 8;
        const float* ap2 = Arow + kb + 16 + khalf * 8;
        *(float4*)(&af[0])  = *(const float4*)(ap);
        *(float4*)(&af[4])  = *(const float4*)(ap + 4);
        *(float4*)(&af[8])  = *(const float4*)(ap2);
        *(float4*)(&af[12]) = *(const float4*)(ap2 + 4);
        v16bf ahi, alo;
        #pragma unroll
        for (int e = 0; e < 16; ++e) { __bf16 h, l; split2(af[e], h, l); ahi[e] = h; alo[e] = l; }

        #pragma unroll
        for (int t = 0; t < 4; ++t) {
            const int j = col_block * 64 + t * 16 + mrow;
            const float* bp = Wm + (size_t)j * KDIM + kb + khalf * 8;
            float bf[16];
            *(float4*)(&bf[0])  = *(const float4*)(bp);
            *(float4*)(&bf[4])  = *(const float4*)(bp + 4);
            *(float4*)(&bf[8])  = *(const float4*)(bp + 16);
            *(float4*)(&bf[12]) = *(const float4*)(bp + 20);
            v16bf bhi, blo;
            #pragma unroll
            for (int e = 0; e < 16; ++e) { __bf16 h, l; split2(bf[e], h, l); bhi[e] = h; blo[e] = l; }

            acc[t] = __builtin_amdgcn_wmma_f32_16x16x32_bf16(false, ahi, false, bhi, (short)0, acc[t], false, false);
            acc[t] = __builtin_amdgcn_wmma_f32_16x16x32_bf16(false, ahi, false, blo, (short)0, acc[t], false, false);
            acc[t] = __builtin_amdgcn_wmma_f32_16x16x32_bf16(false, alo, false, bhi, (short)0, acc[t], false, false);
        }
    }

    const int rowbase = row_tile * 16 + khalf * 8;
    #pragma unroll
    for (int t = 0; t < 4; ++t) {
        const int col = col_block * 64 + t * 16 + mrow;
        const float bv = bias[col];
        #pragma unroll
        for (int r = 0; r < 8; ++r)
            predicts[(size_t)(rowbase + r) * HID + col] = acc[t][r] + bv;
    }
}

// ---------------- per-group logits + logsumexp partial ----------------
__global__ void __launch_bounds__(256)
cpc_logits_kernel(const float* __restrict__ emb,
                  const float* __restrict__ predicts,
                  const int*   __restrict__ neg_perm,
                  float*       __restrict__ partials)
{
    const int i = blockIdx.x;
    const int t = threadIdx.x;

    __shared__ float p_lds[HID];
    __shared__ float red[256];

    p_lds[t] = predicts[(size_t)i * HID + t];
    __syncthreads();

    int idx = neg_perm[(size_t)i * MNEG + t];
    if (idx >= i * 4) idx += 4;
    const float* __restrict__ row = emb + (size_t)idx * HID;
    float acc = 0.f;
    #pragma unroll 4
    for (int l = 0; l < HID; l += 4) {
        float4 r4 = *(const float4*)(row + l);
        acc += p_lds[l]   * r4.x;
        acc += p_lds[l+1] * r4.y;
        acc += p_lds[l+2] * r4.z;
        acc += p_lds[l+3] * r4.w;
    }

    const float hy = emb[((size_t)i * 4 + 3) * HID + t];
    red[t] = p_lds[t] * hy;
    __syncthreads();
    for (int s = 128; s > 0; s >>= 1) { if (t < s) red[t] += red[t + s]; __syncthreads(); }
    const float pos = red[0];
    __syncthreads();

    red[t] = fmaxf(acc, pos);
    __syncthreads();
    for (int s = 128; s > 0; s >>= 1) { if (t < s) red[t] = fmaxf(red[t], red[t + s]); __syncthreads(); }
    const float gmax = red[0];
    __syncthreads();

    red[t] = __expf(acc - gmax);
    __syncthreads();
    for (int s = 128; s > 0; s >>= 1) { if (t < s) red[t] += red[t + s]; __syncthreads(); }

    if (t == 0) {
        const float total = red[0] + __expf(pos - gmax);
        partials[i] = __logf(total) + gmax - pos;
    }
}

// ---------------- deterministic final reduction ----------------
__global__ void __launch_bounds__(256)
cpc_reduce_kernel(const float* __restrict__ partials, float* __restrict__ out)
{
    __shared__ float red[256];
    const int t = threadIdx.x;
    float s = 0.f;
    for (int i = t; i < NGRP; i += 256) s += partials[i];
    red[t] = s;
    __syncthreads();
    for (int st = 128; st > 0; st >>= 1) { if (t < st) red[t] += red[t + st]; __syncthreads(); }
    if (t == 0) out[0] = red[0] * (1.0f / (float)NGRP);
}

extern "C" void kernel_launch(void* const* d_in, const int* in_sizes, int n_in,
                              void* d_out, int out_size, void* d_ws, size_t ws_size,
                              hipStream_t stream) {
    const float* emb      = (const float*)d_in[0];   // (16384, 256) f32
    const float* Wm       = (const float*)d_in[1];   // (256, 768)   f32
    const float* bias     = (const float*)d_in[2];   // (256,)       f32
    const int*   neg_perm = (const int*)  d_in[4];   // (4096, 256)  i32

    const size_t EMB_N = (size_t)NGRP * 4 * HID;     // 4,194,304 elements
    const size_t W_N   = (size_t)HID * KDIM;         //   196,608 elements

    char* ws = (char*)d_ws;
    size_t off = 0;
    float* predicts = (float*)(ws + off); off += (size_t)NGRP * HID * sizeof(float);  // 4 MB
    float* partials = (float*)(ws + off); off += (size_t)NGRP * sizeof(float);        // 16 KB
    size_t off_bf16 = off;
    __bf16* ehi = (__bf16*)(ws + off_bf16); off_bf16 += EMB_N * 2;
    __bf16* elo = (__bf16*)(ws + off_bf16); off_bf16 += EMB_N * 2;
    __bf16* whi = (__bf16*)(ws + off_bf16); off_bf16 += W_N * 2;
    __bf16* wlo = (__bf16*)(ws + off_bf16); off_bf16 += W_N * 2;

    if (ws_size >= off_bf16) {
        // precompute bf16 hi/lo splits once, then pure-bf16 WMMA GEMM
        cpc_split_kernel<<<(int)((EMB_N + 255) / 256), 256, 0, stream>>>(emb, ehi, elo, (int)EMB_N);
        cpc_split_kernel<<<(int)((W_N   + 255) / 256), 256, 0, stream>>>(Wm,  whi, wlo, (int)W_N);
        cpc_gemm_bf16_kernel<<<128, 256, 0, stream>>>(ehi, elo, whi, wlo, bias, predicts);
    } else {
        cpc_gemm_fallback_kernel<<<128, 256, 0, stream>>>(emb, Wm, bias, predicts);
    }

    cpc_logits_kernel<<<NGRP, 256, 0, stream>>>(emb, predicts, neg_perm, partials);
    cpc_reduce_kernel<<<1,    256, 0, stream>>>(partials, (float*)d_out);
}
